// SCLSTM_87780541595939
// MI455X (gfx1250) — matrix operand order repaired and checked by reading, not compiled, not run on hardware
//
#include <hip/hip_runtime.h>
#include <hip/hip_bf16.h>

// ---------------- problem constants ----------------
constexpr int   Bc   = 64;
constexpr int   Tc   = 512;
constexpr int   INc  = 512;
constexpr int   Hc   = 512;
constexpr int   Sc   = 128;
constexpr int   Lc   = 2;
constexpr float ALPHAc = 0.5f;

constexpr int   NWGc  = 64;   // 4 batch tiles x 16 H-slices
constexpr int   WGSZc = 256;  // 8 wave32

typedef __attribute__((ext_vector_type(16))) __bf16    v16bf;
typedef __attribute__((ext_vector_type(8)))  float     v8f;
typedef __attribute__((ext_vector_type(4)))  unsigned  u32x4;

#define GAS __attribute__((address_space(1)))
typedef const unsigned short GAS* gw_t;   // global-addrspace weight pointer

union Frag { v16bf v; u32x4 u[2]; };

__device__ inline unsigned short f2bf(float f) {
    unsigned u = __builtin_bit_cast(unsigned, f);
    unsigned r = u + 0x7FFFu + ((u >> 16) & 1u);   // round-to-nearest-even
    return (unsigned short)(r >> 16);
}

__device__ inline float sigm(float x) { return 1.0f / (1.0f + __expf(-x)); }

__device__ inline v8f wmma_bf16(v16bf a, v16bf b, v8f c) {
    return __builtin_amdgcn_wmma_f32_16x16x32_bf16(
        false, a, false, b, (short)0, c, false, false);
}

// Global-typed + opaque weight base: keeps global_load_b128 (not flat) while
// preventing LLVM from hoisting t-invariant weight fragments across the T-loop
// (which previously caused scratch spills).
__device__ inline gw_t make_global(const unsigned short* p) {
    gw_t g = (gw_t)(uintptr_t)p;
    asm volatile("" : "+s"(g));
    return g;
}

// A fragment: 16(M) x 32(K) bf16, row-major activations (LDS or global).
// lanes 0-15: row = lane, K 0..15; lanes 16-31: row = lane-16, K 16..31
__device__ inline v16bf loadA(const unsigned short* row0, int rowStride, int lane, int k0) {
    int m = lane & 15, half = lane >> 4;
    const unsigned short* p = row0 + m * rowStride + k0 + half * 16;
    Frag f;
    f.u[0] = *(const u32x4*)(p);
    f.u[1] = *(const u32x4*)(p + 8);
    return f.v;
}

// B fragment: 32(K) x 16(N) bf16 from row-major weight W[N][K] in global memory:
// lane holds column n = lane&15, 16 consecutive K at k0 + 16*(lane>>4) -> one contiguous 32B read
__device__ inline v16bf loadB(gw_t w, int rowStride, int n0, int lane, int k0) {
    int n = lane & 15, half = lane >> 4;
    gw_t p = w + (size_t)(n0 + n) * rowStride + k0 + half * 16;
    Frag f;
    f.u[0] = *(const GAS u32x4*)(p);
    f.u[1] = *(const GAS u32x4*)(p + 8);
    return f.v;
}

// Software-pipelined K-loop: prefetch fragment k+1 while wmma consumes k.
template <int NK>
__device__ inline void gemm_seg(v8f& acc,
                                const unsigned short* arow, int astride,
                                gw_t w, int wstride, int ncol, int wk0,
                                int lane)
{
    v16bf a = loadA(arow, astride, lane, 0);
    v16bf b = loadB(w, wstride, ncol, lane, wk0);
    #pragma unroll
    for (int i = 1; i < NK; ++i) {
        v16bf a2 = loadA(arow, astride, lane, i * 32);
        v16bf b2 = loadB(w, wstride, ncol, lane, wk0 + i * 32);
        acc = wmma_bf16(a, b, acc);
        a = a2; b = b2;
    }
    acc = wmma_bf16(a, b, acc);
}

// ---------------- prep: fp32 -> bf16 weight/input conversion + state reset ----------------
__global__ void sclstm_prep(const float* __restrict__ X,  const float* __restrict__ gW,
                            const float* __restrict__ W_ir, const float* __restrict__ W_hr,
                            const float* __restrict__ sW,
                            unsigned short* __restrict__ xbf,  unsigned short* __restrict__ gwbf,
                            unsigned short* __restrict__ wirbf, unsigned short* __restrict__ whrbf,
                            unsigned short* __restrict__ swbf,  unsigned short* __restrict__ hbf,
                            unsigned* __restrict__ ctr)
{
    const long long gid = (long long)blockIdx.x * blockDim.x + threadIdx.x;
    const long long gs  = (long long)gridDim.x * blockDim.x;
    const long long NX  = (long long)Bc * Tc * INc;
    for (long long i = gid; i < NX; i += gs) xbf[i] = f2bf(X[i]);
    const long long NG  = (long long)Lc * 4 * Hc * (INc + Hc);
    for (long long i = gid; i < NG; i += gs) gwbf[i] = f2bf(gW[i]);
    const long long NIR = (long long)Sc * INc;
    for (long long i = gid; i < NIR; i += gs) wirbf[i] = f2bf(W_ir[i]);
    const long long NHR = (long long)Sc * Lc * Hc;
    for (long long i = gid; i < NHR; i += gs) whrbf[i] = f2bf(ALPHAc * W_hr[i]); // fold alpha
    const long long NSW = (long long)Lc * Hc * Sc;
    for (long long i = gid; i < NSW; i += gs) swbf[i] = f2bf(sW[i]);
    const long long NH  = 2ll * Bc * Lc * Hc;   // ping-pong h buffers -> zero (h0 = 0)
    for (long long i = gid; i < NH; i += gs) hbf[i] = 0;
    if (gid == 0) *ctr = 0u;                    // reset device-wide barrier each launch
}

// ---------------- device-wide barrier (64 resident WGs) ----------------
__device__ inline void grid_sync(unsigned* ctr, unsigned* epoch) {
    __threadfence();
    __syncthreads();
    if (threadIdx.x == 0) {
        unsigned target = ++(*epoch);
        target *= (unsigned)NWGc;
        atomicAdd(ctr, 1u);
        while (atomicAdd(ctr, 0u) < target) __builtin_amdgcn_s_sleep(2);
    }
    __syncthreads();
    __threadfence();
}

// ---------------- persistent recurrent kernel ----------------
__global__ __launch_bounds__(WGSZc)
__attribute__((amdgpu_waves_per_eu(1)))           // grid-limited occupancy: use full VGPR budget
void sclstm_persist(
    const unsigned short* __restrict__ xbf,
    const unsigned short* __restrict__ gwbf,
    const unsigned short* __restrict__ wirbf,
    const unsigned short* __restrict__ whrbf,
    const unsigned short* __restrict__ swbf,
    unsigned short* __restrict__ hbf,
    const float* __restrict__ b_ir, const float* __restrict__ b_hr,
    const float* __restrict__ gb,   const float* __restrict__ sb,
    unsigned* __restrict__ ctr,
    float* __restrict__ out_y, float* __restrict__ out_ht,
    float* __restrict__ out_ct, float* __restrict__ out_st)
{
    __shared__ __align__(16) unsigned short sX[16][INc];     // staged x_t tile (16KB, async DMA)
    __shared__ __align__(16) unsigned short sH[16][Lc * Hc]; // staged h_prev tile (32KB, async DMA)
    __shared__ __align__(16) float          sS[16][Sc];      // slot state (fp32), per batch tile
    __shared__ __align__(16) unsigned short sSbf[16][Sc];    // bf16 mirror (A for slot GEMM)
    __shared__ __align__(16) float          sC[Lc][16][32];  // private cell-state chunk
    __shared__ __align__(16) float          sG[4][16][32];   // staged i/f/o/info tiles
    __shared__ __align__(16) float          sSlot[16][32];   // staged slot_info
    __shared__ unsigned s_epoch;

    const int tid   = threadIdx.x;
    const int lane  = tid & 31;
    const int wave  = tid >> 5;
    const int tile  = blockIdx.x >> 4;   // 0..3  -> batch rows [16*tile, 16*tile+16)
    const int slice = blockIdx.x & 15;   // 0..15 -> H columns [32*slice, 32*slice+32)
    const int b0    = tile * 16;
    const int LH    = Lc * Hc;

    for (int i = tid; i < 16 * Sc; i += WGSZc) { (&sS[0][0])[i] = 0.f; (&sSbf[0][0])[i] = 0; }
    for (int i = tid; i < Lc * 16 * 32; i += WGSZc) (&sC[0][0][0])[i] = 0.f;
    if (tid == 0) s_epoch = 0;
    __syncthreads();

    for (int t = 0; t < Tc; ++t) {
        const unsigned short* hPrev = hbf + (size_t)(t & 1) * Bc * LH;
        unsigned short*       hCur  = hbf + (size_t)((t + 1) & 1) * Bc * LH;
        const unsigned short* xRow0 = xbf + ((size_t)b0 * Tc + t) * INc;   // row stride Tc*INc

        // fresh, global-typed weight bases each step -> global_load from L2, no spills, no flat
        gw_t wir = make_global(wirbf);
        gw_t whr = make_global(whrbf);

        // ---------- async-DMA x_t tile (16x512) and h_prev tile (16x1024) into LDS ----------
        {
            for (int c = tid; c < (16 * INc) / 8; c += WGSZc) {   // 1024 x 16B chunks
                int row = c >> 6;
                int col = (c & 63) << 3;
                const unsigned short* gp = xRow0 + (size_t)row * (Tc * INc) + col;
                unsigned ldsOff = (unsigned)(uintptr_t)(void*)&sX[row][col];
                asm volatile("global_load_async_to_lds_b128 %0, %1, off"
                             :: "v"(ldsOff), "v"(gp) : "memory");
            }
            const unsigned short* hRowG = hPrev + (size_t)b0 * LH;
            for (int c = tid; c < (16 * LH) / 8; c += WGSZc) {    // 2048 x 16B chunks
                int row = c >> 7;
                int col = (c & 127) << 3;
                const unsigned short* gp = hRowG + (size_t)row * LH + col;
                unsigned ldsOff = (unsigned)(uintptr_t)(void*)&sH[row][col];
                asm volatile("global_load_async_to_lds_b128 %0, %1, off"
                             :: "v"(ldsOff), "v"(gp) : "memory");
            }
            asm volatile("s_wait_asynccnt 0x0" ::: "memory");
        }
        __syncthreads();

        // ---------- Phase R: reading gate r (full S per WG, sync-free redundancy) ----------
        {
            const int n0 = wave * 16;              // 8 waves x 16 = S = 128 columns
            v8f acc = {0.f, 0.f, 0.f, 0.f, 0.f, 0.f, 0.f, 0.f};
            gemm_seg<16>(acc, &sX[0][0], INc, wir, INc, n0, 0, lane);   // x part  (LDS A)
            gemm_seg<32>(acc, &sH[0][0], LH,  whr, LH,  n0, 0, lane);   // h part  (LDS A, alpha folded)
            const int n = n0 + (lane & 15);
            const float bias = b_ir[n] + ALPHAc * b_hr[n];
            #pragma unroll
            for (int j = 0; j < 8; ++j) {
                int m = (lane >> 4) * 8 + j;
                float r  = sigm(acc[j] + bias);
                float sv = sS[m][n] * r;
                sS[m][n]   = sv;
                sSbf[m][n] = f2bf(sv);
            }
        }
        __syncthreads();

        // ---------- layers ----------
        for (int l = 0; l < Lc; ++l) {
            const unsigned short* curRow0 = (l == 0) ? &sX[0][0] : (hCur + (size_t)b0 * LH);
            const int curStride           = (l == 0) ? INc : LH;
            const unsigned short* hlRow0  = &sH[0][l * Hc];     // h_prev layer-l from LDS
            gw_t gWl = make_global(gwbf + (size_t)l * 4 * Hc * (INc + Hc));
            gw_t sWl = make_global(swbf + (size_t)l * Hc * Sc);

            // gates: wave -> (gate block g, 16-col subtile); N cols = g*H + slice*32 + sub*16
            {
                const int g = wave >> 1, sub = wave & 1;
                const int ncol = g * Hc + slice * 32 + sub * 16;
                v8f acc = {0.f, 0.f, 0.f, 0.f, 0.f, 0.f, 0.f, 0.f};
                gemm_seg<16>(acc, curRow0, curStride, gWl, INc + Hc, ncol, 0,   lane); // cur part
                gemm_seg<16>(acc, hlRow0,  LH,        gWl, INc + Hc, ncol, INc, lane); // h_l part
                const int nl = lane & 15;
                const float gbias = gb[l * 4 * Hc + ncol + nl];
                #pragma unroll
                for (int j = 0; j < 8; ++j) {
                    int m = (lane >> 4) * 8 + j;
                    sG[g][m][sub * 16 + nl] = acc[j] + gbias;
                }
            }

            // slot_info = tanh(s @ sW.T + sb) for this WG's 32-col chunk (waves 0,1)
            if (wave < 2) {
                const int ncolH = slice * 32 + wave * 16;          // global H column
                v8f acc = {0.f, 0.f, 0.f, 0.f, 0.f, 0.f, 0.f, 0.f};
                gemm_seg<4>(acc, &sSbf[0][0], Sc, sWl, Sc, ncolH, 0, lane);    // A from LDS
                const int nl = lane & 15;
                const float sbias = sb[l * Hc + ncolH + nl];
                #pragma unroll
                for (int j = 0; j < 8; ++j) {
                    int m = (lane >> 4) * 8 + j;
                    sSlot[m][wave * 16 + nl] = tanhf(acc[j] + sbias);
                }
            }
            __syncthreads();

            // recombine: c_n, h_n for this WG's 16x32 chunk
            for (int e = tid; e < 512; e += WGSZc) {
                int m = e >> 5, jj = e & 31;
                float ig = sG[0][m][jj], fg = sG[1][m][jj];
                float og = sG[2][m][jj], info = sG[3][m][jj];
                float cO = sC[l][m][jj];
                float cN = sigm(fg) * cO + sigm(ig) * tanhf(info) + sSlot[m][jj];
                float hN = sigm(og) * tanhf(cN);
                sC[l][m][jj] = cN;
                const int b = b0 + m, hc = slice * 32 + jj;
                hCur[(size_t)b * LH + l * Hc + hc] = f2bf(hN);
                if (l == Lc - 1) out_y[((size_t)b * Tc + t) * Hc + hc] = hN;
                if (t == Tc - 1) {
                    out_ht[((size_t)b * Lc + l) * Hc + hc] = hN;
                    out_ct[((size_t)b * Lc + l) * Hc + hc] = cN;
                }
            }
            grid_sync(ctr, &s_epoch);   // publish h_l(new) before anyone consumes it
        }
    }

    // final slot state
    __syncthreads();
    if (slice == 0) {
        for (int e = tid; e < 16 * Sc; e += WGSZc) {
            int m = e >> 7, n = e & (Sc - 1);
            out_st[(size_t)(b0 + m) * Sc + n] = sS[m][n];
        }
    }
}

extern "C" void kernel_launch(void* const* d_in, const int* in_sizes, int n_in,
                              void* d_out, int out_size, void* d_ws, size_t ws_size,
                              hipStream_t stream)
{
    (void)in_sizes; (void)n_in; (void)out_size; (void)ws_size;

    const float* X    = (const float*)d_in[0];
    const float* W_ir = (const float*)d_in[1];
    const float* b_ir = (const float*)d_in[2];
    const float* W_hr = (const float*)d_in[3];
    const float* b_hr = (const float*)d_in[4];
    const float* gW   = (const float*)d_in[5];
    const float* gb   = (const float*)d_in[6];
    const float* sW   = (const float*)d_in[7];
    const float* sb   = (const float*)d_in[8];

    char* ws = (char*)d_ws;
    size_t off = 0;
    auto alloc = [&](size_t bytes) { void* p = ws + off; off += (bytes + 255) & ~(size_t)255; return p; };
    unsigned short* xbf   = (unsigned short*)alloc((size_t)Bc * Tc * INc * 2);
    unsigned short* gwbf  = (unsigned short*)alloc((size_t)Lc * 4 * Hc * (INc + Hc) * 2);
    unsigned short* wirbf = (unsigned short*)alloc((size_t)Sc * INc * 2);
    unsigned short* whrbf = (unsigned short*)alloc((size_t)Sc * Lc * Hc * 2);
    unsigned short* swbf  = (unsigned short*)alloc((size_t)Lc * Hc * Sc * 2);
    unsigned short* hbf   = (unsigned short*)alloc((size_t)2 * Bc * Lc * Hc * 2);
    unsigned*       ctr   = (unsigned*)alloc(256);

    float* out_y  = (float*)d_out;
    float* out_ht = out_y  + (size_t)Bc * Tc * Hc;
    float* out_ct = out_ht + (size_t)Bc * Lc * Hc;
    float* out_st = out_ct + (size_t)Bc * Lc * Hc;

    sclstm_prep<<<2048, 256, 0, stream>>>(X, gW, W_ir, W_hr, sW,
                                          xbf, gwbf, wirbf, whrbf, swbf, hbf, ctr);
    sclstm_persist<<<NWGc, WGSZc, 0, stream>>>(xbf, gwbf, wirbf, whrbf, swbf, hbf,
                                               b_ir, b_hr, gb, sb, ctr,
                                               out_y, out_ht, out_ct, out_st);
}
